// lm_54168127537334
// MI455X (gfx1250) — matrix operand order repaired and checked by reading, not compiled, not run on hardware
//
#include <hip/hip_runtime.h>
#include <hip/hip_bf16.h>
#include <math.h>

typedef __bf16 bf16_t;
typedef __attribute__((ext_vector_type(16))) __bf16 v16bf;
typedef __attribute__((ext_vector_type(8)))  float  v8f;

union FragU { v16bf v; float4 f[2]; };

// A fragment (16x32 bf16, M x K), A row-major with leading dim lda.
// ISA layout: lane m (0-15) holds row m, K = g*8 + {0..7} and 16 + g*8 + {0..7},
// where g = lane>>4.  Two contiguous 16-byte loads.
__device__ __forceinline__ v16bf load_a_frag(const bf16_t* Arow, int lda, int kbase, int lane) {
  int m = lane & 15, g = lane >> 4;
  const bf16_t* p = Arow + (size_t)m * lda + kbase + g * 8;
  FragU u;
  u.f[0] = *(const float4*)p;
  u.f[1] = *(const float4*)(p + 16);
  return u.v;
}

// B fragment (32x16 bf16, K x N) where B is given as Bblk[n][k] (row n of the
// weight matrix = column n of B), contiguous in K.  Lane n=L&15 holds column n,
// K = (L>>4)*8 + {0..7} and 16 + (L>>4)*8 + {0..7}: two 16-byte loads.
__device__ __forceinline__ v16bf load_b_frag(const bf16_t* Bblk, int ldb, int kbase, int lane) {
  int n = lane & 15, g = lane >> 4;
  const bf16_t* p = Bblk + (size_t)n * ldb + kbase + g * 8;
  FragU u;
  u.f[0] = *(const float4*)p;
  u.f[1] = *(const float4*)(p + 16);
  return u.v;
}

__device__ __forceinline__ v8f wmma_bf16(v16bf a, v16bf b, v8f c) {
  return __builtin_amdgcn_wmma_f32_16x16x32_bf16(false, a, false, b, (short)0, c, false, false);
}

// ---------------- small utility kernels ----------------

__global__ void k_f32_to_bf16(const float* __restrict__ src, bf16_t* __restrict__ dst, size_t n) {
  size_t i = (size_t)blockIdx.x * blockDim.x + threadIdx.x;
  size_t stride = (size_t)gridDim.x * blockDim.x;
  for (; i < n; i += stride) dst[i] = (bf16_t)src[i];
}

// s_bf[b][d] = bf16(user_emb[user[b]][d] + item_emb[item[b]][d]),  B=128, MF=256
__global__ void k_gather_user_item(const int* __restrict__ user, const int* __restrict__ item,
                                   const float* __restrict__ uemb, const float* __restrict__ iemb,
                                   bf16_t* __restrict__ s_bf) {
  int i = blockIdx.x * blockDim.x + threadIdx.x;
  int b = i >> 8, d = i & 255;
  s_bf[i] = (bf16_t)(uemb[(size_t)user[b] * 256 + d] + iemb[(size_t)item[b] * 256 + d]);
}

// x_bf[row][d] = bf16(word_emb[tokens[row]][d]),  row = t*B+b, d<512
__global__ void k_gather_words(const int* __restrict__ tokens, const float* __restrict__ wemb,
                               bf16_t* __restrict__ x_bf) {
  int i = blockIdx.x * blockDim.x + threadIdx.x;
  int row = i >> 9, d = i & 511;
  x_bf[i] = (bf16_t)wemb[(size_t)tokens[row] * 512 + d];
}

// ---------------- generic bf16 WMMA GEMM: C[M,N] = A[M,K] @ Bm[N,K]^T + bias ----------------
// grid = (M/16, N/64), block = 128 (4 waves, one 16x16 n-subtile each).
// act==1 -> tanh.  C (fp32) and Cbf (bf16 copy) each optional.
__global__ void __launch_bounds__(128) k_gemm_bf16(
    const bf16_t* __restrict__ A, const bf16_t* __restrict__ Bm,
    const float* __restrict__ bias, float* __restrict__ C, bf16_t* __restrict__ Cbf,
    int K, int N, int act) {
  int lane = threadIdx.x & 31;
  int wave = threadIdx.x >> 5;
  int mt = blockIdx.x;
  int n0 = blockIdx.y * 64 + wave * 16;
  const bf16_t* Arow = A + (size_t)mt * 16 * K;
  const bf16_t* Bblk = Bm + (size_t)n0 * K;
  v8f acc = {};
  for (int k = 0; k < K; k += 32) {
    v16bf af  = load_a_frag(Arow, K, k, lane);
    v16bf bfv = load_b_frag(Bblk, K, k, lane);
    acc = wmma_bf16(af, bfv, acc);
  }
  int n = n0 + (lane & 15);
  float bv = bias ? bias[n] : 0.0f;
  int mbase = mt * 16 + (lane >> 4) * 8;
#pragma unroll
  for (int r = 0; r < 8; ++r) {
    float v = acc[r] + bv;
    if (act == 1) v = tanhf(v);
    size_t idx = (size_t)(mbase + r) * N + n;
    if (C)   C[idx] = v;
    if (Cbf) Cbf[idx] = (bf16_t)v;
  }
}

// ---------------- one GRU timestep ----------------
// gh = h_prev @ W_hh^T (3 gates via 3 WMMA accumulators sharing the A fragment),
// then fused gate math + h update.  grid=(8,8), block=128.
__global__ void __launch_bounds__(128) k_gru_step(
    const bf16_t* __restrict__ hb_prev, const bf16_t* __restrict__ Whh,
    const float* __restrict__ b_hh, const float* __restrict__ gx_t,
    float* __restrict__ h, bf16_t* __restrict__ hb_next, bf16_t* __restrict__ feats_t) {
  const int H = 512;
  int lane = threadIdx.x & 31, wave = threadIdx.x >> 5;
  int mt = blockIdx.x;
  int c0 = blockIdx.y * 64 + wave * 16;
  const bf16_t* Arow = hb_prev + (size_t)mt * 16 * H;
  const bf16_t* Br = Whh + (size_t)c0 * H;
  const bf16_t* Bz = Whh + (size_t)(c0 + H) * H;
  const bf16_t* Bn = Whh + (size_t)(c0 + 2 * H) * H;
  v8f ar = {}, az = {}, an = {};
  for (int k = 0; k < H; k += 32) {
    v16bf af = load_a_frag(Arow, H, k, lane);
    ar = wmma_bf16(af, load_b_frag(Br, H, k, lane), ar);
    az = wmma_bf16(af, load_b_frag(Bz, H, k, lane), az);
    an = wmma_bf16(af, load_b_frag(Bn, H, k, lane), an);
  }
  int n = c0 + (lane & 15);
  float bhr = b_hh[n], bhz = b_hh[n + H], bhn = b_hh[n + 2 * H];
  int mbase = mt * 16 + (lane >> 4) * 8;
#pragma unroll
  for (int r = 0; r < 8; ++r) {
    int row = mbase + r;
    const float* gxr = gx_t + (size_t)row * 1536;
    float rg = 1.0f / (1.0f + __expf(-(gxr[n]         + ar[r] + bhr)));
    float zg = 1.0f / (1.0f + __expf(-(gxr[n + H]     + az[r] + bhz)));
    float ng = tanhf(gxr[n + 2 * H] + rg * (an[r] + bhn));
    size_t hi = (size_t)row * H + n;
    float hv = (1.0f - zg) * ng + zg * h[hi];
    h[hi] = hv;
    hb_next[hi] = (bf16_t)hv;
    feats_t[hi] = (bf16_t)hv;
  }
}

// ---------------- fused output GEMM + online log-softmax NLL ----------------
// One workgroup per 32 feat rows (A staged in LDS, 32 KB), 8 waves sweep all
// 2000 vocab n-tiles.  Per-lane online (max,sumexp) for its 16 (row) slots,
// target-logit capture when the tile covers tgt[row], serial per-row LDS reduce at end.
__global__ void __launch_bounds__(256) k_logits_loss(
    const bf16_t* __restrict__ feats, const bf16_t* __restrict__ Wout,
    const float* __restrict__ out_b, const int* __restrict__ tgt,
    float* __restrict__ out_loss, float* __restrict__ blk_part) {
  const int K = 512, NT = 32000 / 16;
  __shared__ __align__(16) bf16_t Ash[32 * 512];
  __shared__ float rmaxs[32][128];
  __shared__ float rsums[32][128];
  __shared__ float tlog[32];
  __shared__ int   stgt[32];
  __shared__ float rowloss[32];
  int tid = threadIdx.x, lane = tid & 31, wave = tid >> 5;
  int r0 = blockIdx.x * 32;

  {  // stage 32x512 bf16 A tile into LDS
    const float4* src = (const float4*)(feats + (size_t)r0 * K);
    float4* dst = (float4*)Ash;
    for (int i = tid; i < 32 * K / 8; i += 256) dst[i] = src[i];
  }
  if (tid < 32) stgt[tid] = tgt[r0 + tid];
  __syncthreads();

  int   tg[16];
  float rmax[16], rsum[16];
#pragma unroll
  for (int f = 0; f < 2; ++f)
#pragma unroll
    for (int r = 0; r < 8; ++r) {
      int rowl = f * 16 + r + 8 * (lane >> 4);
      tg[f * 8 + r]   = stgt[rowl];
      rmax[f * 8 + r] = -1e30f;
      rsum[f * 8 + r] = 0.0f;
    }

  for (int nt = wave; nt < NT; nt += 8) {
    const bf16_t* Bp = Wout + (size_t)nt * 16 * K;
    if (nt + 8 < NT) __builtin_prefetch(Wout + (size_t)(nt + 8) * 16 * K, 0, 1);
    v8f acc0 = {}, acc1 = {};
    for (int k = 0; k < K; k += 32) {
      v16bf bfv = load_b_frag(Bp, K, k, lane);
      v16bf a0  = load_a_frag(Ash, K, k, lane);
      v16bf a1  = load_a_frag(Ash + 16 * K, K, k, lane);
      acc0 = wmma_bf16(a0, bfv, acc0);
      acc1 = wmma_bf16(a1, bfv, acc1);
    }
    int n = nt * 16 + (lane & 15);
    float bb = out_b[n];
#pragma unroll
    for (int f = 0; f < 2; ++f) {
#pragma unroll
      for (int r = 0; r < 8; ++r) {
        int idx = f * 8 + r;
        float v = (f == 0 ? acc0[r] : acc1[r]) + bb;
        if (n == tg[idx]) tlog[f * 16 + r + 8 * (lane >> 4)] = v;
        float m = rmax[idx];
        if (v <= m) {
          rsum[idx] += __expf(v - m);
        } else {
          rsum[idx] = rsum[idx] * __expf(m - v) + 1.0f;
          rmax[idx] = v;
        }
      }
    }
  }
  int slot = wave * 16 + (lane & 15);
#pragma unroll
  for (int f = 0; f < 2; ++f)
#pragma unroll
    for (int r = 0; r < 8; ++r) {
      int rowl = f * 16 + r + 8 * (lane >> 4);
      rmaxs[rowl][slot] = rmax[f * 8 + r];
      rsums[rowl][slot] = rsum[f * 8 + r];
    }
  __syncthreads();
  if (tid < 32) {
    float gm = -1e30f;
    for (int s = 0; s < 128; ++s) gm = fmaxf(gm, rmaxs[tid][s]);
    float gs = 0.0f;
    for (int s = 0; s < 128; ++s) gs += rsums[tid][s] * __expf(rmaxs[tid][s] - gm);
    float lse  = gm + __logf(gs);
    float nll  = lse - tlog[tid];
    float mask = (stgt[tid] != 0) ? 1.0f : 0.0f;
    float loss = nll * mask;
    out_loss[r0 + tid] = loss;
    rowloss[tid] = loss;
  }
  __syncthreads();
  if (tid == 0) {  // deterministic per-block partial (no float atomics)
    float ls = 0.0f, cs = 0.0f;
    for (int i = 0; i < 32; ++i) {
      ls += rowloss[i];
      cs += (stgt[i] != 0) ? 1.0f : 0.0f;
    }
    blk_part[blockIdx.x * 2]     = ls;
    blk_part[blockIdx.x * 2 + 1] = cs;
  }
}

__global__ void k_finalize(const float* __restrict__ blk_part, float* __restrict__ obj_out, int nblk) {
  if (threadIdx.x == 0 && blockIdx.x == 0) {
    float s = 0.0f, c = 0.0f;
    for (int i = 0; i < nblk; ++i) { s += blk_part[2 * i]; c += blk_part[2 * i + 1]; }
    obj_out[0] = s / fmaxf(c, 1.0f);
  }
}

// ---------------- launch ----------------

extern "C" void kernel_launch(void* const* d_in, const int* in_sizes, int n_in,
                              void* d_out, int out_size, void* d_ws, size_t ws_size,
                              hipStream_t stream) {
  const int*   user          = (const int*)d_in[0];
  const int*   item          = (const int*)d_in[1];
  const int*   review_input  = (const int*)d_in[2];
  const int*   review_target = (const int*)d_in[3];
  const float* user_emb      = (const float*)d_in[4];
  const float* item_emb      = (const float*)d_in[5];
  const float* enc_W         = (const float*)d_in[6];
  const float* enc_b         = (const float*)d_in[7];
  const float* word_emb      = (const float*)d_in[8];
  const float* W_ih          = (const float*)d_in[9];
  const float* W_hh          = (const float*)d_in[10];
  const float* b_ih          = (const float*)d_in[11];
  const float* b_hh          = (const float*)d_in[12];
  const float* out_W         = (const float*)d_in[13];
  const float* out_b         = (const float*)d_in[14];
  float* out = (float*)d_out;

  const int B = 128, T = 50, H = 512, MF = 256, V = 32000, G3 = 1536, TB = T * B;

  char* ws = (char*)d_ws;
  size_t off = 0;
  auto take = [&](size_t bytes) -> char* {
    char* p = ws + off;
    off += (bytes + 255) & ~(size_t)255;
    return p;
  };
  bf16_t* Wih_bf  = (bf16_t*)take((size_t)G3 * H * 2);
  bf16_t* Whh_bf  = (bf16_t*)take((size_t)G3 * H * 2);
  bf16_t* Wout_bf = (bf16_t*)take((size_t)V * H * 2);
  bf16_t* encW_bf = (bf16_t*)take((size_t)H * MF * 2);
  bf16_t* x_bf    = (bf16_t*)take((size_t)TB * H * 2);
  float*  gx      = (float*)take((size_t)TB * G3 * 4);
  float*  h       = (float*)take((size_t)B * H * 4);
  bf16_t* hb0     = (bf16_t*)take((size_t)B * H * 2);
  bf16_t* hb1     = (bf16_t*)take((size_t)B * H * 2);
  bf16_t* feats   = (bf16_t*)take((size_t)TB * H * 2);
  bf16_t* s_bf    = (bf16_t*)take((size_t)B * MF * 2);
  float*  blkp    = (float*)take((size_t)(TB / 32) * 2 * 4);

  // weight precision conversion (one-shot per call; tiny vs main GEMM)
  k_f32_to_bf16<<<512, 256, 0, stream>>>(W_ih, Wih_bf, (size_t)G3 * H);
  k_f32_to_bf16<<<512, 256, 0, stream>>>(W_hh, Whh_bf, (size_t)G3 * H);
  k_f32_to_bf16<<<2048, 256, 0, stream>>>(out_W, Wout_bf, (size_t)V * H);
  k_f32_to_bf16<<<128, 256, 0, stream>>>(enc_W, encW_bf, (size_t)H * MF);

  // encoder: h0 = tanh(s @ enc_W^T + enc_b)
  k_gather_user_item<<<(B * MF) / 256, 256, 0, stream>>>(user, item, user_emb, item_emb, s_bf);
  k_gemm_bf16<<<dim3(B / 16, H / 64), 128, 0, stream>>>(s_bf, encW_bf, enc_b, h, hb0, MF, H, 1);

  // all-timestep input projection: GX = word_emb[tokens] @ W_ih^T + b_ih
  k_gather_words<<<(TB * H) / 256, 256, 0, stream>>>(review_input, word_emb, x_bf);
  k_gemm_bf16<<<dim3(TB / 16, G3 / 64), 128, 0, stream>>>(x_bf, Wih_bf, b_ih, gx, nullptr, H, G3, 0);

  // sequential GRU (ping-pong bf16 h snapshots; h itself stays fp32)
  for (int t = 0; t < T; ++t) {
    bf16_t* hp = (t & 1) ? hb1 : hb0;
    bf16_t* hn = (t & 1) ? hb0 : hb1;
    k_gru_step<<<dim3(B / 16, H / 64), 128, 0, stream>>>(
        hp, Whh_bf, b_hh, gx + (size_t)t * B * G3, h, hn, feats + (size_t)t * B * H);
  }

  // fused vocab GEMM + log-softmax NLL (dominant 210 GFLOP) and final objective
  k_logits_loss<<<TB / 32, 256, 0, stream>>>(feats, Wout_bf, out_b, review_target, out, blkp);
  k_finalize<<<1, 1, 0, stream>>>(blkp, out + TB, TB / 32);
}
